// WindowAttention_9259949490812
// MI455X (gfx1250) — matrix operand order repaired and checked by reading, not compiled, not run on hardware
//
#include <hip/hip_runtime.h>

// ---------------------------------------------------------------------------
// Swin window attention, fused per-window kernel for gfx1250 (MI455X, wave32).
// One workgroup (256 threads = 8 wave32) per window. All matrix math on
// v_wmma_f32_16x16x32_f16; f32 accumulation + f32 softmax; intermediates in LDS.
// ---------------------------------------------------------------------------

typedef __attribute__((ext_vector_type(16))) _Float16 v16h;
typedef __attribute__((ext_vector_type(8)))  float    v8f;

#define NTOK   49          // tokens per window (7x7)
#define NPAD   64          // padded rows (4 tiles of 16)
#define CDIM   128
#define HEADS  4
#define DH     32          // head dim  (== WMMA K for f16)
#define NW     64          // mask windows
#define QKVW   384         // 3*C

// LDS strides (halfs). All even (4B pair alignment); row strides chosen so the
// fragment gather patterns land on distinct banks.
#define LDX    136         // x / attn-output buffer [64][128]
#define LDQ    264         // q|k buffer [64][256]
#define LDV    72          // v-transposed buffer [128][64]
#define LDP    72          // per-wave softmax probs [16][64]

union FragU { v16h v; _Float16 h[16]; };

__device__ __forceinline__ v8f wmma_f16(v16h a, v16h b, v8f c) {
  return __builtin_amdgcn_wmma_f32_16x16x32_f16(
      /*neg_a=*/false, a, /*neg_b=*/false, b,
      /*c_mod=*/(short)0, c, /*reuse_a=*/false, /*reuse_b=*/false);
}

// A fragment: logical A[m][k] = src[m*ld + k]  (row-major activations)
// Layout used here, contiguous k (16-bit A 16x32): lane L -> m = L%16, g = L/16;
// VGPR v holds K = 16g + 2v and +1.
__device__ __forceinline__ v16h load_fragA(const _Float16* src, int ld) {
  const int lane = (int)(threadIdx.x & 31u);
  const int m = lane & 15, g = lane >> 4;
  FragU f;
#pragma unroll
  for (int v = 0; v < 8; ++v) {
    const int kb = (g << 4) + (v << 1);
    unsigned tmp;
    __builtin_memcpy(&tmp, src + m * ld + kb, 4);
    __builtin_memcpy(&f.h[2 * v], &tmp, 4);
  }
  return f.v;
}

// B fragment from transposed source: logical B[k][n] = src[n*ld + k].
// Layout used here, contiguous k (B 32x16): lane L -> n = L%16, g = L/16; VGPR v holds K = 16g+2v, +1.
// Per lane: 16 consecutive halfs -> merges into wide LDS/global loads.
__device__ __forceinline__ v16h load_fragBT(const _Float16* src, int ld) {
  const int lane = (int)(threadIdx.x & 31u);
  const int n = lane & 15, g = lane >> 4;
  FragU f;
#pragma unroll
  for (int v = 0; v < 8; ++v) {
    const int kb = (g << 4) + (v << 1);
    unsigned tmp;
    __builtin_memcpy(&tmp, src + n * ld + kb, 4);
    __builtin_memcpy(&f.h[2 * v], &tmp, 4);
  }
  return f.v;
}

// ---------------------------------------------------------------------------
// Setup kernel 1: weight f32 -> f16 (stays L2-resident afterwards).
// ---------------------------------------------------------------------------
__global__ __launch_bounds__(256) void winattn_convert_w(
    const float* __restrict__ qkv_w, const float* __restrict__ proj_w,
    _Float16* __restrict__ wqkv, _Float16* __restrict__ wproj) {
  const int i = (int)(blockIdx.x * 256 + threadIdx.x);
  if (i < QKVW * CDIM) wqkv[i] = (_Float16)qkv_w[i];
  if (i < CDIM * CDIM) wproj[i] = (_Float16)proj_w[i];
}

// ---------------------------------------------------------------------------
// Setup kernel 2: materialize relative-position bias per head, padded to
// 64x64 with -1e30 on invalid key columns (softmax masking baked in).
// rpb[h][m][n], 4*64*64 f32 = 64 KB, L2-resident.
// ---------------------------------------------------------------------------
__global__ __launch_bounds__(256) void winattn_build_rpb(
    const float* __restrict__ bias_table, const int* __restrict__ rel_idx,
    float* __restrict__ rpb) {
  const int i = (int)(blockIdx.x * 256 + threadIdx.x);  // < 4*64*64
  const int h = i >> 12, m = (i >> 6) & 63, n = i & 63;
  float v;
  if (n >= NTOK)      v = -1e30f;
  else if (m >= NTOK) v = 0.0f;
  else                v = bias_table[rel_idx[m * NTOK + n] * HEADS + h];
  rpb[i] = v;
}

// ---------------------------------------------------------------------------
// Fused window attention.
// ---------------------------------------------------------------------------
__global__ __launch_bounds__(256) void winattn_fused(
    const float* __restrict__ x,          // [4096,49,128]
    const float* __restrict__ qkv_b,      // [384]
    const float* __restrict__ proj_b,     // [128]
    const float* __restrict__ mask,       // [64,49,49]
    const float* __restrict__ rpb,        // [4,64,64] padded rel-pos bias
    const _Float16* __restrict__ wqkv,    // [384,128] f16
    const _Float16* __restrict__ wproj,   // [128,128] f16
    float* __restrict__ out)              // [4096,49,128]
{
  __shared__ _Float16 xs[NPAD * LDX];    // stage1: x (f16); stage3+: attn output
  __shared__ _Float16 qkvs[NPAD * LDQ];  // [64][256]: q(scaled) | k
  __shared__ _Float16 vts[CDIM * LDV];   // v transposed: [d=128][token=64]
  __shared__ _Float16 ps[8 * 16 * LDP];  // per-wave softmax probabilities [16][64]

  const int b    = (int)blockIdx.x;
  const int tid  = (int)threadIdx.x;
  const int lane = tid & 31;
  const int wave = tid >> 5;
  const int g    = lane >> 4;   // 16-lane half
  const int nl   = lane & 15;

  // Warm L2 for the f16 weights (global_prefetch_b8); 256 threads cover 96 KB.
  __builtin_prefetch((const char*)wqkv + tid * 384, 0, 1);
  __builtin_prefetch((const char*)wproj + tid * 128, 0, 1);

  // ---- Stage 1: x -> LDS f16 (vectorized), pad rows [49,64) with zero ------
#pragma unroll
  for (int it = 0; it < 4; ++it) {
    const int e = (tid + it * 256) * 8;        // 8 contiguous elements/thread
    const int r = e >> 7, c = e & 127;
    _Float16 h8[8];
    if (r < NTOK) {
      float f8[8];
      __builtin_memcpy(&f8[0], &x[(b * NTOK + r) * CDIM + c], 16);
      __builtin_memcpy(&f8[4], &x[(b * NTOK + r) * CDIM + c + 4], 16);
#pragma unroll
      for (int j = 0; j < 8; ++j) h8[j] = (_Float16)f8[j];
    } else {
#pragma unroll
      for (int j = 0; j < 8; ++j) h8[j] = (_Float16)0.0f;
    }
    __builtin_memcpy(&xs[r * LDX + c], h8, 16);
  }
  __syncthreads();

  // ---- Stage 2: QKV = X * Wqkv^T + b --------------------------------------
  // Each wave owns 3 output col-tiles (ot = wave*3+j); weight fragments are
  // loaded once per ot and reused across all 4 row-tiles. All fragments for a
  // tile are live in distinct registers before the 4 back-to-back WMMAs.
  // Q (ot<8) pre-scaled by Dh^-0.5, row-major; K (8<=ot<16) row-major;
  // V (ot>=16) stored TRANSPOSED [d][token] with one b128 store.
#pragma unroll
  for (int j = 0; j < 3; ++j) {
    const int ot = wave * 3 + j;
    v16h bw[4];
#pragma unroll
    for (int kt = 0; kt < 4; ++kt)
      bw[kt] = load_fragBT(&wqkv[(ot * 16) * CDIM + kt * 32], CDIM);
    const float bias = qkv_b[ot * 16 + nl];
    const float sc = (ot < 8) ? 0.17677669529663687f : 1.0f;  // 1/sqrt(32)
#pragma unroll
    for (int mt = 0; mt < 4; ++mt) {
      v16h a[4];
#pragma unroll
      for (int kt = 0; kt < 4; ++kt)
        a[kt] = load_fragA(&xs[(mt * 16) * LDX + kt * 32], LDX);
      v8f acc;
#pragma unroll
      for (int r = 0; r < 8; ++r) acc[r] = bias;
#pragma unroll
      for (int kt = 0; kt < 4; ++kt)
        acc = wmma_f16(a[kt], bw[kt], acc);

      const int m0 = mt * 16 + (g << 3);        // token base for this lane
      if (ot < 16) {
        const int c0 = ot * 16 + nl;
#pragma unroll
        for (int r = 0; r < 8; ++r)
          qkvs[(m0 + r) * LDQ + c0] = (_Float16)(acc[r] * sc);
      } else {
        const int d = (ot - 16) * 16 + nl;      // 0..127 output feature
        _Float16 h8[8];
#pragma unroll
        for (int r = 0; r < 8; ++r) h8[r] = (_Float16)acc[r];
        __builtin_memcpy(&vts[d * LDV + m0], h8, 16);  // 8 tokens, contiguous
      }
    }
  }
  __syncthreads();

  // ---- Stage 3: per (head, row-tile): scores -> softmax -> attn*V ----------
  const int w = b & (NW - 1);                   // mask window index
  _Float16* myps = &ps[wave * 16 * LDP];
#pragma unroll
  for (int pi = 0; pi < 2; ++pi) {
    const int pair = wave * 2 + pi;             // 16 pairs over 8 waves
    const int h = pair >> 2, mt = pair & 3;

    // Q fragment (full head dim, K=32) + all four K fragments up-front.
    v16h qf = load_fragA(&qkvs[(mt * 16) * LDQ + h * DH], LDQ);
    v16h kf[4];
#pragma unroll
    for (int nt = 0; nt < 4; ++nt)
      kf[nt] = load_fragBT(&qkvs[(nt * 16) * LDQ + CDIM + h * DH], LDQ);

    v8f s[4];
#pragma unroll
    for (int nt = 0; nt < 4; ++nt) {
      // Accumulator pre-init: padded rel-pos bias (incl. -1e30 masking) plus
      // the attention mask; both coalesced b32 loads, no index gather.
#pragma unroll
      for (int r = 0; r < 8; ++r) {
        const int m = mt * 16 + (g << 3) + r;
        const int n = nt * 16 + nl;
        float init = rpb[((h << 6) + m) * 64 + n];
        if (m < NTOK && n < NTOK)
          init += mask[(w * NTOK + m) * NTOK + n];
        s[nt][r] = init;
      }
      s[nt] = wmma_f16(qf, kf[nt], s[nt]);
    }

    // Softmax over 64 cols: 4 regs/lane + xor-shuffle across the 16-lane half
    // that owns the row (masks 1/2/4/8 keep lane/16 invariant on wave32).
#pragma unroll
    for (int r = 0; r < 8; ++r) {
      float mx = fmaxf(fmaxf(s[0][r], s[1][r]), fmaxf(s[2][r], s[3][r]));
#pragma unroll
      for (int off = 1; off <= 8; off <<= 1)
        mx = fmaxf(mx, __shfl_xor(mx, off, 32));
      float e[4], sum = 0.0f;
#pragma unroll
      for (int nt = 0; nt < 4; ++nt) {
        e[nt] = __expf(s[nt][r] - mx);
        sum += e[nt];
      }
#pragma unroll
      for (int off = 1; off <= 8; off <<= 1)
        sum += __shfl_xor(sum, off, 32);
      const float inv = 1.0f / sum;
      const int ml = (g << 3) + r;              // row within this 16-row tile
#pragma unroll
      for (int nt = 0; nt < 4; ++nt)
        myps[ml * LDP + nt * 16 + nl] = (_Float16)(e[nt] * inv);
    }

    // Cross-lane LDS RAW (P written by all lanes, fragments gathered by all
    // lanes of the same wave): drain DScnt before the A-fragment loads.
    asm volatile("s_wait_dscnt 0x0" ::: "memory");

    // O = P(16x64) * V(64x32): V^T in LDS -> paired BT fragments.
    v16h af[2];
#pragma unroll
    for (int kt = 0; kt < 2; ++kt)
      af[kt] = load_fragA(&myps[kt * 32], LDP);
#pragma unroll
    for (int dt = 0; dt < 2; ++dt) {
      v16h vf[2];
#pragma unroll
      for (int kt = 0; kt < 2; ++kt)
        vf[kt] = load_fragBT(&vts[(h * DH + dt * 16) * LDV + kt * 32], LDV);
      v8f o = {};
#pragma unroll
      for (int kt = 0; kt < 2; ++kt)
        o = wmma_f16(af[kt], vf[kt], o);
      const int c0 = h * DH + dt * 16 + nl;
      const int m0 = mt * 16 + (g << 3);
#pragma unroll
      for (int r = 0; r < 8; ++r)
        xs[(m0 + r) * LDX + c0] = (_Float16)o[r];   // xs reused as attn output
    }
    asm volatile("s_wait_dscnt 0x0" ::: "memory");  // before ps reuse next pair
  }
  __syncthreads();

  // ---- Stage 4: out = O * Wproj^T + b; store valid rows as f32 -------------
  // Each wave owns one output col-tile; proj-weight fragments loaded once and
  // reused across all 4 row-tiles.
  {
    const int ot = wave;
    v16h bw[4];
#pragma unroll
    for (int kt = 0; kt < 4; ++kt)
      bw[kt] = load_fragBT(&wproj[(ot * 16) * CDIM + kt * 32], CDIM);
    const float bias = proj_b[ot * 16 + nl];
#pragma unroll
    for (int mt = 0; mt < 4; ++mt) {
      v16h a[4];
#pragma unroll
      for (int kt = 0; kt < 4; ++kt)
        a[kt] = load_fragA(&xs[(mt * 16) * LDX + kt * 32], LDX);
      v8f acc;
#pragma unroll
      for (int r = 0; r < 8; ++r) acc[r] = bias;
#pragma unroll
      for (int kt = 0; kt < 4; ++kt)
        acc = wmma_f16(a[kt], bw[kt], acc);

      const int m0 = mt * 16 + (g << 3);
      const int c  = ot * 16 + nl;
#pragma unroll
      for (int r = 0; r < 8; ++r) {
        const int m = m0 + r;
        if (m < NTOK) out[(b * NTOK + m) * CDIM + c] = acc[r];
      }
    }
  }
}

// ---------------------------------------------------------------------------
extern "C" void kernel_launch(void* const* d_in, const int* in_sizes, int n_in,
                              void* d_out, int out_size, void* d_ws, size_t ws_size,
                              hipStream_t stream) {
  const float* x          = (const float*)d_in[0];
  const float* qkv_w      = (const float*)d_in[1];
  const float* qkv_b      = (const float*)d_in[2];
  const float* proj_w     = (const float*)d_in[3];
  const float* proj_b     = (const float*)d_in[4];
  const float* bias_table = (const float*)d_in[5];
  const int*   rel_idx    = (const int*)d_in[6];
  const float* mask       = (const float*)d_in[7];
  float* out = (float*)d_out;

  // Workspace layout: f16 weights + padded per-head rel-pos bias.
  _Float16* wqkv  = (_Float16*)d_ws;                  // 384*128 halfs (96 KB)
  _Float16* wproj = wqkv + QKVW * CDIM;               // 128*128 halfs (32 KB)
  float*    rpb   = (float*)(wproj + CDIM * CDIM);    // 4*64*64 f32   (64 KB)

  winattn_convert_w<<<(QKVW * CDIM + 255) / 256, 256, 0, stream>>>(
      qkv_w, proj_w, wqkv, wproj);
  winattn_build_rpb<<<(HEADS * 64 * 64) / 256, 256, 0, stream>>>(
      bias_table, rel_idx, rpb);

  // One workgroup per window.
  winattn_fused<<<4096, 256, 0, stream>>>(
      x, qkv_b, proj_b, mask, rpb, wqkv, wproj, out);
}
